// VisionTransformer_37976100831909
// MI455X (gfx1250) — matrix operand order repaired and checked by reading, not compiled, not run on hardware
//
#include <hip/hip_runtime.h>

// ---------------------------------------------------------------------------
// ViT-Base forward for MI455X (gfx1250, wave32, WMMA + async LDS staging).
// All large GEMMs + attention run through v_wmma_f32_16x16x32_bf16.
// ---------------------------------------------------------------------------

typedef __attribute__((ext_vector_type(16))) __bf16 v16bf;
typedef __attribute__((ext_vector_type(8)))  float  v8f;

#define D_MODEL 768
#define FF_DIM  3072
#define NBATCH  32
#define SEQ     197
#define NHEAD   12
#define NLAYER  12
#define MROWS   (NBATCH * SEQ)      // 6304
#define MPATCH  (NBATCH * 196)      // 6272
#define SKP     224                 // padded key count (7 tiles of 32)

#if defined(__HIP_DEVICE_COMPILE__)
#if __has_builtin(__builtin_amdgcn_global_load_async_to_lds_b128) && \
    __has_builtin(__builtin_amdgcn_s_wait_asynccnt)
#define HAVE_ASYNC_LDS 1
#endif
#endif

#ifdef HAVE_ASYNC_LDS
typedef int v4i_vs __attribute__((vector_size(16)));
typedef __attribute__((address_space(1))) v4i_vs* gas1_v4p;
typedef __attribute__((address_space(3))) v4i_vs* las3_v4p;

__device__ __forceinline__ void async_copy_b128(const void* gsrc, void* ldst) {
  __builtin_amdgcn_global_load_async_to_lds_b128(
      (gas1_v4p)gsrc, (las3_v4p)ldst, 0, 0);
}
#endif

__device__ __forceinline__ unsigned short f32_to_bf16(float f) {
  unsigned int u = __float_as_uint(f);
  u += 0x7FFFu + ((u >> 16) & 1u);            // round-to-nearest-even
  return (unsigned short)(u >> 16);
}

__device__ __forceinline__ v8f zero8() {
  v8f z;
#pragma unroll
  for (int i = 0; i < 8; ++i) z[i] = 0.f;
  return z;
}

// A-fragment (16x32 bf16) from LDS, row-major rows of `stride` bf16.
// Layout used here (contiguous k, not ISA table order): lane m=L&15, half=L>>4; VGPR r holds K = half*16+2*r, +1
__device__ __forceinline__ v16bf load_a_frag(const unsigned short* base, int stride, int lane) {
  int m = lane & 15, half = lane >> 4;
  union { v16bf v; unsigned int u[8]; } f;
  const unsigned short* rowp = base + m * stride;
#pragma unroll
  for (int r = 0; r < 8; ++r) {
    int k = half * 16 + (r << 1);
    f.u[r] = *(const unsigned int*)(rowp + k);
  }
  return f.v;
}

// B-fragment (32x16 bf16) from LDS stored transposed: [n][k], k contiguous.
// Layout used here (contiguous k, same as load_a_frag): lane n=L&15, half=L>>4; 16 contiguous K values = half*16 + 0..15
__device__ __forceinline__ v16bf load_b_frag(const unsigned short* base, int stride, int lane) {
  int n = lane & 15, half = lane >> 4;
  union { v16bf v; unsigned int u[8]; } f;
  const unsigned int* colp = (const unsigned int*)(base + n * stride + half * 16);
#pragma unroll
  for (int r = 0; r < 8; ++r) f.u[r] = colp[r];
  return f.v;
}

__device__ __forceinline__ v8f wmma_bf16(v16bf a, v16bf b, v8f c) {
  return __builtin_amdgcn_wmma_f32_16x16x32_bf16(false, a, false, b, (short)0, c, false, false);
}

// ---------------------------------------------------------------------------
// Generic GEMM: C[M,N] = act( A_bf16[M,K](lda) @ W_f32[K,N] + bias )
// Block: 256 threads (8 waves), tile 128x128, K-step 32.
// Waves arranged 2x4; each wave computes 64x32 (4x2 fragments, 8 WMMAs/K-step).
// A tile staged via GLOBAL_LOAD_ASYNC_TO_LDS_B128 when available.
// ---------------------------------------------------------------------------
#define BM 128
#define BN 128
#define BK 32
#define NT 256

__global__ void __launch_bounds__(NT)
gemm_bf16_kernel(const unsigned short* __restrict__ A, long lda,
                 const float* __restrict__ W, const float* __restrict__ bias,
                 float* __restrict__ Cf, unsigned short* __restrict__ Cb,
                 int M, int K, int N, int relu)
{
  __shared__ unsigned short As[BM * BK];   // [128 rows][32 k], row-major bf16 (8KB)
  __shared__ unsigned short Wt[BN * BK];   // transposed: [128 n][32 k] bf16  (8KB)

  int t    = threadIdx.x;
  int lane = t & 31;
  int wave = t >> 5;
  int wr   = (wave >> 2) * 64;             // wave row offset inside tile (0/64)
  int wc   = (wave & 3) * 32;              // wave col offset inside tile (0..96)
  int tileM = blockIdx.y * BM;
  int tileN = blockIdx.x * BN;

  v8f acc[4][2];
#pragma unroll
  for (int i = 0; i < 4; ++i)
#pragma unroll
    for (int j = 0; j < 2; ++j) acc[i][j] = zero8();

  const bool fullM = (tileM + BM) <= M;

  for (int k0 = 0; k0 < K; k0 += BK) {
    // --- stage A tile (bf16 global -> LDS), 512 x 16B segments, 2 per thread
#ifdef HAVE_ASYNC_LDS
    if (fullM) {
#pragma unroll
      for (int i = t; i < (BM * BK) / 8; i += NT) {
        int row = i >> 2;
        int seg = i & 3;
        async_copy_b128(A + (long)(tileM + row) * lda + k0 + seg * 8, As + i * 8);
      }
    } else
#endif
    {
#pragma unroll
      for (int i = t; i < (BM * BK) / 8; i += NT) {
        int row = i >> 2;
        int seg = i & 3;
        int gr  = tileM + row;
        uint4 val = make_uint4(0u, 0u, 0u, 0u);
        if (gr < M)
          val = *(const uint4*)(A + (long)gr * lda + k0 + seg * 8);
        ((uint4*)As)[i] = val;
      }
    }
    // --- stage W tile (f32 global -> bf16 LDS, transposed), 16 per thread
#pragma unroll
    for (int i = t; i < BK * BN; i += NT) {
      int r = i >> 7;                       // k row 0..31
      int n = i & 127;
      float w = 0.f;
      if (tileN + n < N) w = W[(long)(k0 + r) * N + tileN + n];
      Wt[n * BK + r] = f32_to_bf16(w);
    }
    if (k0 + BK < K)
      __builtin_prefetch(&W[(long)(k0 + BK) * N + tileN], 0, 1);
#ifdef HAVE_ASYNC_LDS
    __builtin_amdgcn_s_wait_asynccnt(0);
#endif
    __syncthreads();

    v16bf af[4], bf[2];
#pragma unroll
    for (int fi = 0; fi < 4; ++fi)
      af[fi] = load_a_frag(As + (wr + fi * 16) * BK, BK, lane);
#pragma unroll
    for (int fj = 0; fj < 2; ++fj)
      bf[fj] = load_b_frag(Wt + (wc + fj * 16) * BK, BK, lane);

#pragma unroll
    for (int fi = 0; fi < 4; ++fi)
#pragma unroll
      for (int fj = 0; fj < 2; ++fj)
        acc[fi][fj] = wmma_bf16(af[fi], bf[fj], acc[fi][fj]);
    __syncthreads();
  }

  // --- epilogue: bias, relu, store f32 and/or bf16
  int half = lane >> 4, ln = lane & 15;
#pragma unroll
  for (int fi = 0; fi < 4; ++fi) {
#pragma unroll
    for (int fj = 0; fj < 2; ++fj) {
      int col = tileN + wc + fj * 16 + ln;
      if (col >= N) continue;
      float bv = bias ? bias[col] : 0.f;
#pragma unroll
      for (int r = 0; r < 8; ++r) {
        int row = tileM + wr + fi * 16 + half * 8 + r;
        if (row >= M) continue;
        float v = acc[fi][fj][r] + bv;
        if (relu) v = fmaxf(v, 0.f);
        if (Cf) Cf[(long)row * N + col] = v;
        if (Cb) Cb[(long)row * N + col] = f32_to_bf16(v);
      }
    }
  }
}

// ---------------------------------------------------------------------------
// Attention: one wave per (q-strip of 16, head, batch).
// scores = (Q Kᵀ)/8 -> softmax -> AV, all contractions via WMMA.
// ---------------------------------------------------------------------------
__global__ void __launch_bounds__(32)
attention_kernel(const unsigned short* __restrict__ Q,
                 const unsigned short* __restrict__ Km,
                 const unsigned short* __restrict__ V,
                 unsigned short* __restrict__ AV)
{
  __shared__ unsigned short Qs[16 * 64];   // Q strip [16 q][64]
  __shared__ unsigned short KVs[64 * 32];  // K tile [32 key][64]  OR  V tile [64 d][32 key]
  __shared__ float          Sc[16 * SKP];  // raw scores / exp
  __shared__ unsigned short Ps[16 * SKP];  // probs bf16

  int lane = threadIdx.x;
  int qs = blockIdx.x;           // 0..12
  int hh = blockIdx.y;           // head
  int b  = blockIdx.z;           // batch
  int qBase = qs * 16;
  long rowBase = (long)b * SEQ * D_MODEL + hh * 64;   // element offset of (b, s=0, head col)

  // stage Q strip (bf16, zero-padded rows) -- 512 uints
  for (int i = lane; i < 512; i += 32) {
    int row = i >> 5, c = i & 31;
    int s = qBase + row;
    unsigned int val = 0u;
    if (s < SEQ) val = ((const unsigned int*)(Q + rowBase + (long)s * D_MODEL))[c];
    ((unsigned int*)Qs)[i] = val;
  }
  __syncthreads();

  v16bf a0 = load_a_frag(Qs, 64, lane);       // k-dim 0..31
  v16bf a1 = load_a_frag(Qs + 32, 64, lane);  // k-dim 32..63
  int half = lane >> 4, ln = lane & 15;

  // ---- scores ----
  for (int kt = 0; kt < 7; ++kt) {
    for (int i = lane; i < 1024; i += 32) {   // K tile, 1024 uints = [32 key][64]
      int key = i >> 5, c = i & 31;
      int ks = kt * 32 + key;
      unsigned int val = 0u;
      if (ks < SEQ) val = ((const unsigned int*)(Km + rowBase + (long)ks * D_MODEL))[c];
      ((unsigned int*)KVs)[i] = val;
    }
    __syncthreads();
#pragma unroll
    for (int g = 0; g < 2; ++g) {
      v16bf bf0 = load_b_frag(KVs + g * 16 * 64,      64, lane);
      v16bf bf1 = load_b_frag(KVs + g * 16 * 64 + 32, 64, lane);
      v8f c = zero8();
      c = wmma_bf16(a0, bf0, c);
      c = wmma_bf16(a1, bf1, c);
      int col = kt * 32 + g * 16 + ln;
#pragma unroll
      for (int r = 0; r < 8; ++r)
        Sc[(half * 8 + r) * SKP + col] = c[r] * 0.125f;   // 1/sqrt(64)
    }
    __syncthreads();
  }

  // ---- softmax: two lanes per row ----
  {
    int row = lane & 15, sub = lane >> 4;
    float mx = -1e30f;
    for (int c = sub; c < SEQ; c += 2) mx = fmaxf(mx, Sc[row * SKP + c]);
    mx = fmaxf(mx, __shfl_xor(mx, 16, 32));
    float sum = 0.f;
    for (int c = sub; c < SKP; c += 2) {
      float e = (c < SEQ) ? __expf(Sc[row * SKP + c] - mx) : 0.f;
      Sc[row * SKP + c] = e;
      sum += e;
    }
    sum += __shfl_xor(sum, 16, 32);
    float inv = 1.f / sum;
    for (int c = sub; c < SKP; c += 2)
      Ps[row * SKP + c] = f32_to_bf16(Sc[row * SKP + c] * inv);
  }
  __syncthreads();

  // ---- AV = probs @ V ----
  v8f accv[4];
#pragma unroll
  for (int g = 0; g < 4; ++g) accv[g] = zero8();

  for (int kt = 0; kt < 7; ++kt) {
    for (int i = lane; i < 2048; i += 32) {   // V tile transposed: [64 d][32 key]
      int key = i >> 6, d = i & 63;
      int ks = kt * 32 + key;
      unsigned short val = 0;
      if (ks < SEQ) val = V[rowBase + (long)ks * D_MODEL + d];
      KVs[d * 32 + key] = val;
    }
    __syncthreads();
    v16bf ap = load_a_frag(Ps + kt * 32, SKP, lane);
#pragma unroll
    for (int g = 0; g < 4; ++g) {
      v16bf bv = load_b_frag(KVs + g * 16 * 32, 32, lane);
      accv[g] = wmma_bf16(ap, bv, accv[g]);
    }
    __syncthreads();
  }

#pragma unroll
  for (int g = 0; g < 4; ++g)
#pragma unroll
    for (int r = 0; r < 8; ++r) {
      int s = qBase + half * 8 + r;
      if (s < SEQ)
        AV[rowBase + (long)s * D_MODEL + g * 16 + ln] = f32_to_bf16(accv[g][r]);
    }
}

// ---------------------------------------------------------------------------
// Patchify: x[B,3,224,224] -> Xp_bf16[B*196, 768], feature order (c, ph, pw)
// ---------------------------------------------------------------------------
__global__ void patchify_kernel(const float* __restrict__ x, unsigned short* __restrict__ Xp)
{
  long idx = (long)blockIdx.x * blockDim.x + threadIdx.x;
  if (idx >= (long)MPATCH * D_MODEL) return;
  int f = (int)(idx % D_MODEL);
  long p = idx / D_MODEL;
  int b  = (int)(p / 196);
  int pp = (int)(p % 196);
  int gy = pp / 14, gx = pp % 14;
  int c  = f >> 8;
  int py = (f >> 4) & 15;
  int px = f & 15;
  float v = x[((long)(b * 3 + c) * 224 + (gy * 16 + py)) * 224 + gx * 16 + px];
  Xp[idx] = f32_to_bf16(v);
}

// cls + pos:  h[b,0,:] = cls + pos[0];  h[b,s,:] = P[b,s-1,:] + pos[s]
__global__ void addpos_kernel(const float* __restrict__ P, const float* __restrict__ cls,
                              const float* __restrict__ pos,
                              float* __restrict__ Hf, unsigned short* __restrict__ Hb)
{
  long idx = (long)blockIdx.x * blockDim.x + threadIdx.x;
  if (idx >= (long)MROWS * D_MODEL) return;
  int d = (int)(idx % D_MODEL);
  long r = idx / D_MODEL;
  int s = (int)(r % SEQ);
  int b = (int)(r / SEQ);
  float v;
  if (s == 0) v = cls[d] + pos[d];
  else        v = P[((long)b * 196 + (s - 1)) * D_MODEL + d] + pos[(long)s * D_MODEL + d];
  Hf[idx] = v;
  Hb[idx] = f32_to_bf16(v);
}

// h += LayerNorm(T)*g + b; also refresh bf16 mirror. One block per row.
__global__ void __launch_bounds__(256)
ln_add_kernel(const float* __restrict__ T, const float* __restrict__ gam,
              const float* __restrict__ bet,
              float* __restrict__ Hf, unsigned short* __restrict__ Hb)
{
  __shared__ float red[256];
  long row = blockIdx.x;
  const float* tr = T + row * D_MODEL;
  int t = threadIdx.x;

  float x0 = tr[t], x1 = tr[t + 256], x2 = tr[t + 512];
  red[t] = x0 + x1 + x2;
  __syncthreads();
  for (int off = 128; off > 0; off >>= 1) {
    if (t < off) red[t] += red[t + off];
    __syncthreads();
  }
  float mu = red[0] * (1.f / 768.f);
  __syncthreads();

  float d0 = x0 - mu, d1 = x1 - mu, d2 = x2 - mu;
  red[t] = d0 * d0 + d1 * d1 + d2 * d2;
  __syncthreads();
  for (int off = 128; off > 0; off >>= 1) {
    if (t < off) red[t] += red[t + off];
    __syncthreads();
  }
  float rs = rsqrtf(red[0] * (1.f / 768.f) + 1e-5f);

#pragma unroll
  for (int i = 0; i < 3; ++i) {
    int c = t + i * 256;
    float dv = (i == 0 ? d0 : (i == 1 ? d1 : d2));
    float v = Hf[row * D_MODEL + c] + dv * rs * gam[c] + bet[c];
    Hf[row * D_MODEL + c] = v;
    Hb[row * D_MODEL + c] = f32_to_bf16(v);
  }
}

// ---------------------------------------------------------------------------
// Host orchestration
// ---------------------------------------------------------------------------
extern "C" void kernel_launch(void* const* d_in, const int* in_sizes, int n_in,
                              void* d_out, int out_size, void* d_ws, size_t ws_size,
                              hipStream_t stream)
{
  const float* x        = (const float*)d_in[0];
  const float* W_patch  = (const float*)d_in[1];
  const float* b_patch  = (const float*)d_in[2];
  const float* cls_tok  = (const float*)d_in[3];
  const float* pos      = (const float*)d_in[4];
  const float* Wq       = (const float*)d_in[5];
  const float* bq       = (const float*)d_in[6];
  const float* Wk       = (const float*)d_in[7];
  const float* bk       = (const float*)d_in[8];
  const float* Wv       = (const float*)d_in[9];
  const float* bv       = (const float*)d_in[10];
  const float* Wo       = (const float*)d_in[11];
  const float* bo       = (const float*)d_in[12];
  const float* ln1_g    = (const float*)d_in[13];
  const float* ln1_b    = (const float*)d_in[14];
  const float* W1       = (const float*)d_in[15];
  const float* b1       = (const float*)d_in[16];
  const float* W2       = (const float*)d_in[17];
  const float* b2       = (const float*)d_in[18];
  const float* ln2_g    = (const float*)d_in[19];
  const float* ln2_b    = (const float*)d_in[20];
  const float* W_head   = (const float*)d_in[21];
  const float* b_head   = (const float*)d_in[22];

  char* ws = (char*)d_ws;
  auto carve = [&](size_t bytes) -> void* {
    void* p = (void*)ws;
    ws += (bytes + 255) & ~(size_t)255;
    return p;
  };

  unsigned short* Xp  = (unsigned short*)carve((size_t)MPATCH * D_MODEL * 2);
  float*          Hf  = (float*)carve((size_t)MROWS * D_MODEL * 4);
  unsigned short* Hb  = (unsigned short*)carve((size_t)MROWS * D_MODEL * 2);
  unsigned short* Qb  = (unsigned short*)carve((size_t)MROWS * D_MODEL * 2);
  unsigned short* Kb  = (unsigned short*)carve((size_t)MROWS * D_MODEL * 2);
  unsigned short* Vb  = (unsigned short*)carve((size_t)MROWS * D_MODEL * 2);
  unsigned short* AVb = (unsigned short*)carve((size_t)MROWS * D_MODEL * 2);
  float*          Tmp = (float*)carve((size_t)MROWS * D_MODEL * 4);
  unsigned short* F1b = (unsigned short*)carve((size_t)MROWS * FF_DIM * 2);

  // 1) patchify
  {
    long n = (long)MPATCH * D_MODEL;
    patchify_kernel<<<dim3((unsigned)((n + 255) / 256)), dim3(256), 0, stream>>>(x, Xp);
  }
  // 2) patch projection -> Tmp (f32)
  gemm_bf16_kernel<<<dim3((D_MODEL + BN - 1) / BN, (MPATCH + BM - 1) / BM), dim3(NT), 0, stream>>>(
      Xp, (long)D_MODEL, W_patch, b_patch, Tmp, nullptr, MPATCH, D_MODEL, D_MODEL, 0);
  // 3) cls + pos -> Hf/Hb
  {
    long n = (long)MROWS * D_MODEL;
    addpos_kernel<<<dim3((unsigned)((n + 255) / 256)), dim3(256), 0, stream>>>(Tmp, cls_tok, pos, Hf, Hb);
  }

  dim3 gemmBlk(NT);
  dim3 grid768((D_MODEL + BN - 1) / BN, (MROWS + BM - 1) / BM);
  dim3 gridFF((FF_DIM + BN - 1) / BN, (MROWS + BM - 1) / BM);

  for (int l = 0; l < NLAYER; ++l) {
    const long wOff = (long)l * D_MODEL * D_MODEL;
    const long bOff = (long)l * D_MODEL;
    // QKV projections (bf16 outputs)
    gemm_bf16_kernel<<<grid768, gemmBlk, 0, stream>>>(
        Hb, (long)D_MODEL, Wq + wOff, bq + bOff, nullptr, Qb, MROWS, D_MODEL, D_MODEL, 0);
    gemm_bf16_kernel<<<grid768, gemmBlk, 0, stream>>>(
        Hb, (long)D_MODEL, Wk + wOff, bk + bOff, nullptr, Kb, MROWS, D_MODEL, D_MODEL, 0);
    gemm_bf16_kernel<<<grid768, gemmBlk, 0, stream>>>(
        Hb, (long)D_MODEL, Wv + wOff, bv + bOff, nullptr, Vb, MROWS, D_MODEL, D_MODEL, 0);
    // attention
    attention_kernel<<<dim3(13, NHEAD, NBATCH), dim3(32), 0, stream>>>(Qb, Kb, Vb, AVb);
    // output projection -> Tmp (f32)
    gemm_bf16_kernel<<<grid768, gemmBlk, 0, stream>>>(
        AVb, (long)D_MODEL, Wo + wOff, bo + bOff, Tmp, nullptr, MROWS, D_MODEL, D_MODEL, 0);
    // h += LN(Tmp)
    ln_add_kernel<<<dim3(MROWS), dim3(256), 0, stream>>>(Tmp, ln1_g + bOff, ln1_b + bOff, Hf, Hb);
    // FFN1 (relu) -> F1b (bf16)
    gemm_bf16_kernel<<<gridFF, gemmBlk, 0, stream>>>(
        Hb, (long)D_MODEL, W1 + (long)l * D_MODEL * FF_DIM, b1 + (long)l * FF_DIM,
        nullptr, F1b, MROWS, D_MODEL, FF_DIM, 1);
    // FFN2 -> Tmp (f32)
    gemm_bf16_kernel<<<grid768, gemmBlk, 0, stream>>>(
        F1b, (long)FF_DIM, W2 + (long)l * FF_DIM * D_MODEL, b2 + bOff,
        Tmp, nullptr, MROWS, FF_DIM, D_MODEL, 0);
    // h += LN(Tmp)
    ln_add_kernel<<<dim3(MROWS), dim3(256), 0, stream>>>(Tmp, ln2_g + bOff, ln2_b + bOff, Hf, Hb);
  }

  // head: CLS rows (stride S*D in Hb) @ W_head + b_head -> d_out [32, 768] f32
  gemm_bf16_kernel<<<dim3((D_MODEL + BN - 1) / BN, 1), gemmBlk, 0, stream>>>(
      Hb, (long)SEQ * D_MODEL, W_head, b_head, (float*)d_out, nullptr,
      NBATCH, D_MODEL, D_MODEL, 0);
}
